// DSRB_50740743635180
// MI455X (gfx1250) — matrix operand order repaired and checked by reading, not compiled, not run on hardware
//
#include <hip/hip_runtime.h>

// ---------------------------------------------------------------------------
// DSRB spiking dense block for MI455X (gfx1250, wave32, WMMA).
//
//   copy x -> feats[ch 0..63]
//   for l in 0..3:
//     lif(feats[:, :Cin]) -> spikes (bf16 bits, 0/1 exact)
//     conv3x3 via v_wmma_f32_16x16x32_bf16 (implicit im2col GEMM) -> raw f32
//     bn stats + apply -> feats[ch 64+24l ..]
//   lif(feats[:, :160]) -> spikes
//   conv1x1 WMMA (same kernel, ksize=1, grid.z tiles N=64) -> convout
//   attention tail + residual -> d_out
//
// Conv kernel: block = 256 threads (8 waves), macro tile 128(M) x 32(N);
// one block == one output image row (W=128). Full bf16 B panel + per-k
// im2col decode tables resident in dynamic LDS (~97KB; CDNA5 WGP has 320KB).
// Fragments are 2x ds_load_b128 each; C stored as 2x float4 per acc.
// Requires ws_size >= ~352 MB.
// ---------------------------------------------------------------------------

typedef __attribute__((ext_vector_type(16))) __bf16          v16bf;
typedef __attribute__((ext_vector_type(8)))  float           v8f;
typedef __attribute__((ext_vector_type(8)))  unsigned short  us8;

#define HWD   16384   // H*W
#define WD    128
#define NIMG  16      // T*B
#define CST   160     // feats/spikes channel stride
#define TT    4
#define BB    4

#define KMAX      1248          // max padded Ktot (136*9=1224 -> 1248)
#define BP_STRIDE 1240          // u16; 2480B rows, 16B aligned; >= max Ktot 1224
#define AS_STRIDE 40            // u16; 80B rows, 16B aligned
#define AS_U16    (128 * AS_STRIDE)                 //  5120 u16
#define BP_U16    (32 * BP_STRIDE)                  // 39680 u16
#define ROWB_OFF  ((AS_U16 + BP_U16) * 2)           // byte offset of int rowb[KMAX]
#define DWT_OFF   (ROWB_OFF + KMAX * 4)             // byte offset of short dwt[KMAX]
#define CONV_LDS  (DWT_OFF + KMAX * 2)              // 97,088 B dynamic LDS
#define ROW_OOB   (-(1 << 28))                      // sentinel: input row out of bounds

static __device__ __forceinline__ unsigned short f2bf(float f) {
  union { float f; unsigned int u; } c; c.f = f;
  unsigned int u = c.u + 0x7FFFu + ((c.u >> 16) & 1u);   // round-to-nearest-even
  return (unsigned short)(u >> 16);
}
static __device__ __forceinline__ float sigm(float x) {
  return 1.0f / (1.0f + __expf(-x));
}

// ---------------- weights f32 -> bf16 bits (once) ---------------------------
__global__ void w2bf_kernel(const float* __restrict__ src,
                            unsigned short* __restrict__ dst, int n) {
  int i = blockIdx.x * 256 + threadIdx.x;
  if (i < n) dst[i] = f2bf(src[i]);
}

// ---------------- copy x into feats channels [0,64) -------------------------
__global__ void copy_x_kernel(const float* __restrict__ x, float* __restrict__ feats) {
  size_t idx = (size_t)blockIdx.x * 256 + threadIdx.x;    // over 16*64*HW
  if (idx >= (size_t)NIMG * 64 * HWD) return;
  int i   = idx & (HWD - 1);
  int c   = (idx >> 14) & 63;
  int img = idx >> 20;
  feats[((size_t)img * CST + c) * HWD + i] = x[idx];
}

// ---------------- multi-step LIF (tau=2, vth=0.15, hard reset) --------------
__global__ void lif_kernel(const float* __restrict__ feats,
                           unsigned short* __restrict__ spikes, int Cin) {
  size_t idx = (size_t)blockIdx.x * 256 + threadIdx.x;    // over B*Cin*HW
  if (idx >= (size_t)BB * Cin * HWD) return;
  int i = idx & (HWD - 1);
  int c = (int)((idx >> 14) % Cin);
  int b = (int)(idx / ((size_t)Cin << 14));
  float v = 0.f;
#pragma unroll
  for (int t = 0; t < TT; ++t) {
    size_t p = ((size_t)(t * BB + b) * CST + c) * HWD + i;
    v = v * 0.5f + feats[p];                               // decay_input=False
    if (v - 0.15f >= 0.f) { spikes[p] = 0x3F80; v = 0.f; } // spike + hard reset
    else                  { spikes[p] = 0;                }
  }
}

// ---------------- implicit-GEMM conv via v_wmma_f32_16x16x32_bf16 ----------
// grid = (128 rows, 16 imgs, N/32); block = 256 = 8 waves.
// wave w owns M rows [w*16, w*16+16) of this image row; computes 2 N-subtiles.
__global__ void wconv_kernel(const unsigned short* __restrict__ spikes, // [16][CST][HW] bf16 bits
                             const unsigned short* __restrict__ wbf,    // [Cout][Ktot] bf16 bits
                             float* __restrict__ outraw,                // [16][Cout][HW]
                             int Cin, int Cout, int ksize, int pad) {
  extern __shared__ unsigned short smem[];
  unsigned short* As   = smem;                       // [128][AS_STRIDE]
  unsigned short* Bp   = smem + AS_U16;              // [32][BP_STRIDE]
  int*            rowb = (int*)((char*)smem + ROWB_OFF);  // [KMAX]
  short*          dwt  = (short*)((char*)smem + DWT_OFF); // [KMAX]

  const int tid  = threadIdx.x;
  const int img  = blockIdx.y;
  const int h    = blockIdx.x;                     // output row (M = h*128 + wc)
  const int n_base = blockIdx.z * 32;
  const int wave = tid >> 5, lane = tid & 31;
  const int khalf = lane >> 4;
  const int Ktot = Cin * ksize * ksize;            // always a multiple of 8
  const int Kpad = (Ktot + 31) & ~31;
  const unsigned short* simg = spikes + (size_t)img * CST * HWD;

  __builtin_prefetch(wbf + (size_t)n_base * Ktot, 0, 1);   // global_prefetch_b8

  // ---- one-time per block: im2col decode tables ----
  // rowb[k] = ci*HW + ih*W + dw  (ih = h + kh - pad), or ROW_OOB if row invalid
  // element address = simg[rowb[k] + mi]; column valid iff 0 <= mi+dwt[k] < W
  for (int k = tid; k < Kpad; k += 256) {
    int v = ROW_OOB; short dw = 0;
    if (k < Ktot) {
      int ci, kh, kw2;
      if (ksize == 3) { ci = k / 9; int r = k - ci * 9; kh = r / 3; kw2 = r - kh * 3; }
      else            { ci = k; kh = 0; kw2 = 0; }
      int ih = h + kh - pad;
      dw = (short)(kw2 - pad);
      if (ih >= 0 && ih < WD) v = ci * HWD + ih * WD + dw;
    }
    rowb[k] = v; dwt[k] = dw;
  }

  // ---- stage full B panel once: Bp[n][k] = wbf[n_base+n][k] (b128 copies) ----
  const int K8 = Kpad >> 3;
  for (int n = 0; n < 32; ++n) {
    const int ng = n_base + n;
    const us8* srcrow = (const us8*)(wbf + (size_t)ng * Ktot);
    for (int j = tid; j < K8; j += 256) {
      us8 v = {};
      if (ng < Cout && (j << 3) < Ktot) v = srcrow[j];
      *(us8*)&Bp[n * BP_STRIDE + (j << 3)] = v;
    }
  }
  __syncthreads();

  v8f acc0 = {}, acc1 = {};
  for (int k0 = 0; k0 < Ktot; k0 += 32) {
    // ---- stage A: table-driven im2col gather, coalesced along mi ----
    // 2048 tasks = 16 kk-pairs x 128 mi; 8 per thread; packed ds_store_b32.
#pragma unroll
    for (int ii = 0; ii < 8; ++ii) {
      int task = ii * 256 + tid;
      int mi = task & 127, kk = (task >> 7) << 1;
      unsigned int p = 0;
#pragma unroll
      for (int q = 0; q < 2; ++q) {
        int k = k0 + kk + q;                       // k < Kpad always
        int rb = rowb[k];
        int iw = mi + dwt[k];
        unsigned short v = 0;
        if (rb != ROW_OOB && (unsigned)iw < (unsigned)WD)
          v = simg[rb + mi];
        p |= (unsigned int)v << (16 * q);
      }
      *(unsigned int*)&As[mi * AS_STRIDE + kk] = p;
    }
    __syncthreads();

    // ---- fragments: 2x ds_load_b128 each, contiguous k = 16*khalf + s in A and B ----
    union { v16bf v; us8 h[2]; } fa, fb0, fb1;
    const unsigned short* arow = &As[(wave * 16 + (lane & 15)) * AS_STRIDE];
    fa.h[0] = *(const us8*)(arow + khalf * 16);
    fa.h[1] = *(const us8*)(arow + khalf * 16 + 8);
    const int coln = lane & 15;
    const int kbb  = k0 + khalf * 16;
    fb0.h[0] = *(const us8*)&Bp[coln * BP_STRIDE + kbb];
    fb0.h[1] = *(const us8*)&Bp[coln * BP_STRIDE + kbb + 8];
    fb1.h[0] = *(const us8*)&Bp[(16 + coln) * BP_STRIDE + kbb];
    fb1.h[1] = *(const us8*)&Bp[(16 + coln) * BP_STRIDE + kbb + 8];

    acc0 = __builtin_amdgcn_wmma_f32_16x16x32_bf16(false, fa.v, false, fb0.v,
                                                   (short)0, acc0, false, false);
    acc1 = __builtin_amdgcn_wmma_f32_16x16x32_bf16(false, fa.v, false, fb1.v,
                                                   (short)0, acc1, false, false);
    __syncthreads();
  }

  // ---- store C: per lane 8 consecutive M floats -> 2x float4 per acc ----
  const int moff = h * WD + wave * 16 + 8 * khalf;
  const int n0 = n_base + (lane & 15);
  if (n0 < Cout) {
    float* dst = outraw + ((size_t)img * Cout + n0) * HWD + moff;
    *(float4*)(dst)     = make_float4(acc0[0], acc0[1], acc0[2], acc0[3]);
    *(float4*)(dst + 4) = make_float4(acc0[4], acc0[5], acc0[6], acc0[7]);
  }
  const int n1 = n0 + 16;
  if (n1 < Cout) {
    float* dst = outraw + ((size_t)img * Cout + n1) * HWD + moff;
    *(float4*)(dst)     = make_float4(acc1[0], acc1[1], acc1[2], acc1[3]);
    *(float4*)(dst + 4) = make_float4(acc1[4], acc1[5], acc1[6], acc1[7]);
  }
}

// ---------------- BN: per-channel mean & gamma*rsqrt(var+eps) ---------------
__global__ void bn_stats_kernel(const float* __restrict__ raw,
                                const float* __restrict__ gamma,
                                float* __restrict__ mean_out,
                                float* __restrict__ scale_out, int Cout) {
  const int c = blockIdx.x;
  __shared__ float ssum[256], ssq[256];
  float s = 0.f, q = 0.f;
  for (int i = threadIdx.x; i < NIMG * HWD; i += 256) {
    int img = i >> 14, off = i & (HWD - 1);
    float y = raw[((size_t)img * Cout + c) * HWD + off];
    s += y; q += y * y;
  }
  ssum[threadIdx.x] = s; ssq[threadIdx.x] = q;
  __syncthreads();
  for (int st = 128; st > 0; st >>= 1) {
    if (threadIdx.x < st) { ssum[threadIdx.x] += ssum[threadIdx.x + st];
                            ssq[threadIdx.x]  += ssq[threadIdx.x + st]; }
    __syncthreads();
  }
  if (threadIdx.x == 0) {
    const float n = (float)NIMG * (float)HWD;
    float m = ssum[0] / n;
    float v = ssq[0] / n - m * m;
    mean_out[c]  = m;
    scale_out[c] = gamma[c] * rsqrtf(v + 1e-5f);
  }
}

__global__ void bn_apply_kernel(const float* __restrict__ raw,
                                const float* __restrict__ mean,
                                const float* __restrict__ scale,
                                const float* __restrict__ beta,
                                float* __restrict__ feats, int Cout, int ch_off) {
  size_t idx = (size_t)blockIdx.x * 256 + threadIdx.x;     // over 16*Cout*HW
  size_t total = (size_t)NIMG * Cout * HWD;
  if (idx >= total) return;
  int i   = idx & (HWD - 1);
  int c   = (int)((idx >> 14) % Cout);
  int img = (int)(idx / ((size_t)Cout << 14));
  float y = raw[idx];
  feats[((size_t)img * CST + ch_off + c) * HWD + i] = (y - mean[c]) * scale[c] + beta[c];
}

// ---------------- attention ------------------------------------------------
__global__ void pool_bc_kernel(const float* __restrict__ convout,
                               float* __restrict__ pooled0) {
  const int bc = blockIdx.x;               // b*64 + c
  const int b = bc >> 6, c = bc & 63;
  __shared__ float ssum[256];
  float s = 0.f;
  for (int i = threadIdx.x; i < TT * HWD; i += 256) {
    int t = i >> 14, off = i & (HWD - 1);
    s += convout[((size_t)(t * BB + b) * 64 + c) * HWD + off];
  }
  ssum[threadIdx.x] = s;
  __syncthreads();
  for (int st = 128; st > 0; st >>= 1) {
    if (threadIdx.x < st) ssum[threadIdx.x] += ssum[threadIdx.x + st];
    __syncthreads();
  }
  if (threadIdx.x == 0) pooled0[bc] = ssum[0] / ((float)TT * (float)HWD);
}

__global__ void mlp_kernel(const float* __restrict__ pooled0,
                           const float* __restrict__ t_w, const float* __restrict__ t_b,
                           const float* __restrict__ c_w1, const float* __restrict__ c_b1,
                           const float* __restrict__ c_w2, const float* __restrict__ c_b2,
                           float* __restrict__ temp_out, float* __restrict__ ca_out) {
  __shared__ float s_p0[256], s_temp[BB], s_h[BB * 4];
  const int tid = threadIdx.x;                 // 64 threads
#pragma unroll
  for (int q = 0; q < 4; ++q) s_p0[tid * 4 + q] = pooled0[tid * 4 + q];
  __syncthreads();
  if (tid < BB) {
    float m = 0.f;
    for (int c = 0; c < 64; ++c) m += s_p0[tid * 64 + c];
    m *= (1.f / 64.f);
    s_temp[tid] = sigm(t_w[0] * m + t_b[0]);
  }
  __syncthreads();
  if (tid < BB * 4) {
    int b = tid >> 2, j = tid & 3;
    float a = c_b1[j];
    for (int c = 0; c < 64; ++c) a += s_temp[b] * s_p0[b * 64 + c] * c_w1[j * 64 + c];
    s_h[tid] = a > 0.f ? a : 0.f;
  }
  __syncthreads();
#pragma unroll
  for (int q = 0; q < 4; ++q) {
    int idx = tid + q * 64;                    // over B*64
    int b = idx >> 6, c = idx & 63;
    float a = c_b2[c];
#pragma unroll
    for (int j = 0; j < 4; ++j) a += s_h[b * 4 + j] * c_w2[c * 4 + j];
    ca_out[idx] = sigm(a);
  }
  if (tid < BB) temp_out[tid] = s_temp[tid];
}

__global__ void sp_kernel(const float* __restrict__ convout,
                          const float* __restrict__ temp, const float* __restrict__ ca,
                          float* __restrict__ sp) {
  int idx = blockIdx.x * 256 + threadIdx.x;    // over B*T*HW, layout [b][t][hw]
  if (idx >= BB * TT * HWD) return;
  int i = idx & (HWD - 1);
  int t = (idx >> 14) & 3;
  int b = idx >> 16;
  float a = 0.f;
#pragma unroll 4
  for (int c = 0; c < 64; ++c)
    a += convout[((size_t)(t * BB + b) * 64 + c) * HWD + i] * ca[b * 64 + c];
  sp[idx] = temp[b] * a * (1.f / 64.f);
}

__global__ void sa_kernel(const float* __restrict__ sp,
                          const float* __restrict__ s_w, const float* __restrict__ s_b,
                          float* __restrict__ sa) {
  int idx = blockIdx.x * 256 + threadIdx.x;    // over B*T*HW
  if (idx >= BB * TT * HWD) return;
  int i = idx & (HWD - 1);
  int h = i >> 7, wc = i & (WD - 1);
  int base = idx & ~(HWD - 1);
  float a = s_b[0];
#pragma unroll
  for (int kh = 0; kh < 3; ++kh) {
#pragma unroll
    for (int kw = 0; kw < 3; ++kw) {
      int ih = h + kh - 1, iw = wc + kw - 1;
      if (ih >= 0 && ih < WD && iw >= 0 && iw < WD)
        a += s_w[kh * 3 + kw] * sp[base + ih * WD + iw];
    }
  }
  sa[idx] = sigm(a);
}

__global__ void final_kernel(const float* __restrict__ convout,
                             const float* __restrict__ x,
                             const float* __restrict__ temp,
                             const float* __restrict__ ca,
                             const float* __restrict__ sa,
                             float* __restrict__ out) {
  size_t idx = (size_t)blockIdx.x * 256 + threadIdx.x;  // over T*B*64*HW
  if (idx >= (size_t)NIMG * 64 * HWD) return;
  int i   = idx & (HWD - 1);
  int c   = (idx >> 14) & 63;
  int img = idx >> 20;                 // t*B + b
  int t = img >> 2, b = img & 3;
  out[idx] = convout[idx] * temp[b] * ca[b * 64 + c]
           * sa[((b * TT + t) << 14) + i] + x[idx];
}

// ---------------------------------------------------------------------------
extern "C" void kernel_launch(void* const* d_in, const int* in_sizes, int n_in,
                              void* d_out, int out_size, void* d_ws, size_t ws_size,
                              hipStream_t stream) {
  (void)in_sizes; (void)n_in; (void)out_size; (void)ws_size;
  const float* x     = (const float*)d_in[0];
  const float* w[4]  = {(const float*)d_in[1], (const float*)d_in[4],
                        (const float*)d_in[7], (const float*)d_in[10]};
  const float* g[4]  = {(const float*)d_in[2], (const float*)d_in[5],
                        (const float*)d_in[8], (const float*)d_in[11]};
  const float* be[4] = {(const float*)d_in[3], (const float*)d_in[6],
                        (const float*)d_in[9], (const float*)d_in[12]};
  const float* lff_w = (const float*)d_in[13];
  const float* t_w   = (const float*)d_in[14];
  const float* t_b   = (const float*)d_in[15];
  const float* c_w1  = (const float*)d_in[16];
  const float* c_b1  = (const float*)d_in[17];
  const float* c_w2  = (const float*)d_in[18];
  const float* c_b2  = (const float*)d_in[19];
  const float* s_w   = (const float*)d_in[20];
  const float* s_b   = (const float*)d_in[21];
  float* out = (float*)d_out;

  // ---- workspace carve (256B aligned) ----
  char* ws = (char*)d_ws;
  size_t off = 0;
  auto carve = [&](size_t bytes) { void* p = ws + off; off = (off + bytes + 255) & ~(size_t)255; return p; };
  float*          feats   = (float*)         carve((size_t)NIMG * CST * HWD * 4); // 168 MB
  unsigned short* spikes  = (unsigned short*)carve((size_t)NIMG * CST * HWD * 2); //  84 MB
  float*          raw     = (float*)         carve((size_t)NIMG * 24  * HWD * 4); //  25 MB
  float*          convout = (float*)         carve((size_t)NIMG * 64  * HWD * 4); //  67 MB
  unsigned short* wbf[4];
  int             kw_n[4];
  for (int l = 0; l < 4; ++l) {
    const int Cin = 64 + 24 * l;
    kw_n[l] = 24 * Cin * 9;
    wbf[l] = (unsigned short*)carve((size_t)kw_n[l] * 2);
  }
  unsigned short* wbf_lff = (unsigned short*)carve((size_t)64 * 160 * 2);
  float*          bn_mean = (float*)carve(64 * 4);
  float*          bn_scal = (float*)carve(64 * 4);
  float*          pooled0 = (float*)carve(256 * 4);
  float*          temp    = (float*)carve(BB * 4);
  float*          ca      = (float*)carve(256 * 4);
  float*          sp      = (float*)carve((size_t)BB * TT * HWD * 4);
  float*          sa      = (float*)carve((size_t)BB * TT * HWD * 4);

  const size_t nfull = (size_t)NIMG * 64 * HWD;           // 16.7M

  // weight conversion (once per call; cheap)
  for (int l = 0; l < 4; ++l)
    w2bf_kernel<<<(kw_n[l] + 255) / 256, 256, 0, stream>>>(w[l], wbf[l], kw_n[l]);
  w2bf_kernel<<<(64 * 160 + 255) / 256, 256, 0, stream>>>(lff_w, wbf_lff, 64 * 160);

  copy_x_kernel<<<(nfull + 255) / 256, 256, 0, stream>>>(x, feats);

  for (int l = 0; l < 4; ++l) {
    const int Cin = 64 + 24 * l;
    const size_t nlif = (size_t)BB * Cin * HWD;
    lif_kernel<<<(nlif + 255) / 256, 256, 0, stream>>>(feats, spikes, Cin);
    wconv_kernel<<<dim3(WD, NIMG, 1), 256, CONV_LDS, stream>>>(
        spikes, wbf[l], raw, Cin, 24, 3, 1);
    bn_stats_kernel<<<24, 256, 0, stream>>>(raw, g[l], bn_mean, bn_scal, 24);
    const size_t nbn = (size_t)NIMG * 24 * HWD;
    bn_apply_kernel<<<(nbn + 255) / 256, 256, 0, stream>>>(
        raw, bn_mean, bn_scal, be[l], feats, 24, 64 + 24 * l);
  }

  // final LIF over all 160 channels + 1x1 conv (GEMM K=160, N=64)
  lif_kernel<<<((size_t)BB * CST * HWD + 255) / 256, 256, 0, stream>>>(feats, spikes, CST);
  wconv_kernel<<<dim3(WD, NIMG, 2), 256, CONV_LDS, stream>>>(
      spikes, wbf_lff, convout, CST, 64, 1, 0);

  // attention
  pool_bc_kernel<<<256, 256, 0, stream>>>(convout, pooled0);
  mlp_kernel<<<1, 64, 0, stream>>>(pooled0, t_w, t_b, c_w1, c_b1, c_w2, c_b2, temp, ca);
  sp_kernel<<<(BB * TT * HWD + 255) / 256, 256, 0, stream>>>(convout, temp, ca, sp);
  sa_kernel<<<(BB * TT * HWD + 255) / 256, 256, 0, stream>>>(sp, s_w, s_b, sa);
  final_kernel<<<(nfull + 255) / 256, 256, 0, stream>>>(convout, x, temp, ca, sa, out);
}
